// MambaMixer_72078141162020
// MI455X (gfx1250) — matrix operand order repaired and checked by reading, not compiled, not run on hardware
//
#include <hip/hip_runtime.h>
#include <hip/hip_bf16.h>
#include <math.h>
#include <stdint.h>

typedef __attribute__((ext_vector_type(16))) __bf16 v16bf;
typedef __attribute__((ext_vector_type(8)))  __bf16 v8bf;
typedef __attribute__((ext_vector_type(4)))  __bf16 v4bf;
typedef __attribute__((ext_vector_type(8)))  float  v8f;

#define L_SEQ 2048
#define H_DIM 2048
#define E_DIM 4096
#define N_SSM 16
#define R_DT  128
#define SSM_LD 256   // x_proj output padded from 160 -> 256 columns

// ---------------------------------------------------------------------------
// CDNA5 async global->LDS DMA (ASYNCcnt), per CDNA5 ISA.
// vdst = per-lane LDS byte offset, vaddr = per-lane 64-bit global address.
// ---------------------------------------------------------------------------
__device__ __forceinline__ void async_load_b128(uint32_t lds_off, const void* gaddr)
{
    asm volatile("global_load_async_to_lds_b128 %0, %1, off"
                 :: "v"(lds_off), "v"(gaddr)
                 : "memory");
}

__device__ __forceinline__ void wait_async0()
{
    asm volatile("s_wait_asynccnt 0" ::: "memory");
}

// low 32 bits of a generic pointer to __shared__ = LDS byte offset
__device__ __forceinline__ uint32_t lds_off_of(const void* p)
{
    return (uint32_t)(uintptr_t)p;
}

// ---------------------------------------------------------------------------
// fp32 -> bf16 bulk convert, 4 elements/thread (b128 load, b64 store).
// ---------------------------------------------------------------------------
__global__ void cvt_f32_bf16(const float* __restrict__ src,
                             __bf16* __restrict__ dst, int n)
{
    int i4 = (blockIdx.x * blockDim.x + threadIdx.x) * 4;
    if (i4 >= n) return;
    float4 v = *(const float4*)(src + i4);
    v4bf p;
    p[0] = (__bf16)v.x; p[1] = (__bf16)v.y;
    p[2] = (__bf16)v.z; p[3] = (__bf16)v.w;
    *(v4bf*)(dst + i4) = p;
}

// x_proj_w [160, E] -> bf16 padded [256, E] (rows 160..255 = 0)
__global__ void padcvt_wx(const float* __restrict__ src, __bf16* __restrict__ dst)
{
    int col = blockIdx.x * blockDim.x + threadIdx.x;
    int row = blockIdx.y;
    float v = (row < R_DT + 2 * N_SSM) ? src[(size_t)row * E_DIM + col] : 0.0f;
    dst[(size_t)row * E_DIM + col] = (__bf16)v;
}

// ---------------------------------------------------------------------------
// GEMM: C[M,N] = A[M,Kd] * B[N,Kd]^T, bf16 in / fp32 out, no bounds checks:
// M % 128 == 0, N % 128 == 0, Kd % 32 == 0, lda/ldb % 8 == 0.
// 256 threads = 8 waves; workgroup tile 128x128; wave tile 32x64
// (2x4 fragments of v_wmma_f32_16x16x32_bf16).
// K-tiles streamed with async global->LDS DMA into double-buffered LDS;
// the buffers are 8KB apart, so the DMA destination toggles with one XOR.
// ---------------------------------------------------------------------------
#define LDS_BUF_STRIDE 8192u    // bytes between buffer 0 and buffer 1

__launch_bounds__(256)
__global__ void gemm_bt_wmma(const __bf16* __restrict__ A, int lda,
                             const __bf16* __restrict__ B, int ldb,
                             float* __restrict__ C, int ldc, int Kd)
{
    __shared__ __align__(16) __bf16 As[2][128][32];   // 2 x 8KB
    __shared__ __align__(16) __bf16 Bs[2][128][32];   // 2 x 8KB

    const int tid   = threadIdx.x;
    const int lane  = tid & 31;
    const int wave  = tid >> 5;
    const int waveM = wave >> 1;            // 0..3 -> 32-row slab
    const int waveN = wave & 1;             // 0..1 -> 64-col slab
    const int half  = lane >> 4;            // half-wave (ISA K sub-chunk select)
    const int l15   = lane & 15;
    const int kA    = half * 8;

    const int blockM = blockIdx.y * 128;
    const int blockN = blockIdx.x * 128;

    v8f acc[2][4] = {};

    // Tile load map: 128 rows x 32 cols bf16 = 512 chunks of 8 bf16; 2/thread.
    // Prologue: DMA K-tile 0 into buffer 0; set up toggling DMA state.
    uint32_t dstA[2], dstB[2];              // next DMA dests (LDS byte offsets)
    const __bf16 *pA[2], *pB[2];            // next DMA sources
    #pragma unroll
    for (int i = 0; i < 2; ++i) {
        int f  = tid + i * 256;
        int r  = f >> 2;
        int c8 = (f & 3) * 8;
        uint32_t a0 = lds_off_of(&As[0][r][c8]);
        uint32_t b0 = lds_off_of(&Bs[0][r][c8]);
        const __bf16* ga = A + (size_t)(blockM + r) * lda + c8;
        const __bf16* gb = B + (size_t)(blockN + r) * ldb + c8;
        async_load_b128(a0, ga);
        async_load_b128(b0, gb);
        dstA[i] = a0 ^ LDS_BUF_STRIDE;      // next tile -> buffer 1
        dstB[i] = b0 ^ LDS_BUF_STRIDE;
        pA[i] = ga + 32;
        pB[i] = gb + 32;
    }

    const int nIter = Kd >> 5;
    for (int it = 0; it < nIter; ++it) {
        const int buf = it & 1;

        wait_async0();          // this wave's DMA for tile `it` has landed
        __syncthreads();        // all waves: tile `it` visible, other buf free

        if (it + 1 < nIter) {   // DMA next tile while WMMA consumes this one
            #pragma unroll
            for (int i = 0; i < 2; ++i) {
                async_load_b128(dstA[i], pA[i]);
                async_load_b128(dstB[i], pB[i]);
                dstA[i] ^= LDS_BUF_STRIDE;
                dstB[i] ^= LDS_BUF_STRIDE;
                pA[i] += 32;
                pB[i] += 32;
            }
        }

        // B fragments (32x16, K x N): n = l15; lanes 0-15 K=0..7,16..23; 16-31 K=8..15,24..31
        v16bf bfrag[4];
        #pragma unroll
        for (int nt = 0; nt < 4; ++nt) {
            int br = waveN * 64 + nt * 16 + l15;
            v8bf blo = *(const v8bf*)&Bs[buf][br][half * 8];
            v8bf bhi = *(const v8bf*)&Bs[buf][br][16 + half * 8];
            bfrag[nt] = __builtin_shufflevector(blo, bhi,
                0, 1, 2, 3, 4, 5, 6, 7, 8, 9, 10, 11, 12, 13, 14, 15);
        }

        #pragma unroll
        for (int mt = 0; mt < 2; ++mt) {
            // A fragment 16x32: all lanes hold M = l15; half selects K sub-chunk
            int ar = waveM * 32 + mt * 16 + l15;
            v8bf alo = *(const v8bf*)&As[buf][ar][kA];
            v8bf ahi = *(const v8bf*)&As[buf][ar][16 + kA];
            v16bf afrag = __builtin_shufflevector(alo, ahi,
                0, 1, 2, 3, 4, 5, 6, 7, 8, 9, 10, 11, 12, 13, 14, 15);
            #pragma unroll
            for (int nt = 0; nt < 4; ++nt) {
                acc[mt][nt] = __builtin_amdgcn_wmma_f32_16x16x32_bf16(
                    false, afrag, false, bfrag[nt],
                    (short)0, acc[mt][nt], false, false);
            }
        }
    }

    // C layout: VGPR r -> M = r + 8*half, N = l15
    #pragma unroll
    for (int mt = 0; mt < 2; ++mt) {
        #pragma unroll
        for (int nt = 0; nt < 4; ++nt) {
            int gm0 = blockM + waveM * 32 + mt * 16 + half * 8;
            int gn  = blockN + waveN * 64 + nt * 16 + l15;
            #pragma unroll
            for (int r = 0; r < 8; ++r)
                C[(size_t)(gm0 + r) * ldc + gn] = acc[mt][nt][r];
        }
    }
}

// ---------------------------------------------------------------------------
// Depthwise causal conv1d (K=4, left pad 3) + bias + SiLU.
// Emits fp32 (for the scan) and bf16 (for the x_proj GEMM).
// ---------------------------------------------------------------------------
__global__ void mamba_conv_silu(const float* __restrict__ proj,   // [L,2E]
                                const float* __restrict__ cw,     // [E,1,4]
                                const float* __restrict__ cb,     // [E]
                                float* __restrict__ h,            // [L,E]
                                __bf16* __restrict__ hbf)         // [L,E]
{
    int e = blockIdx.x * blockDim.x + threadIdx.x;
    int l = blockIdx.y;
    float acc = cb[e];
    #pragma unroll
    for (int k = 0; k < 4; ++k) {
        int l2 = l - 3 + k;
        if (l2 >= 0)
            acc = fmaf(proj[(size_t)l2 * (2 * E_DIM) + e], cw[e * 4 + k], acc);
    }
    float sv = acc / (1.0f + __expf(-acc));   // SiLU
    h[(size_t)l * E_DIM + e]   = sv;
    hbf[(size_t)l * E_DIM + e] = (__bf16)sv;
}

// ---------------------------------------------------------------------------
// Selective scan: one lane per channel e; N=16 states in registers.
// Fuses softplus(dt+bias), recurrence, y = sum_n s_n*C_n, D-skip, SiLU gate.
// ---------------------------------------------------------------------------
__launch_bounds__(256)
__global__ void mamba_scan(const float* __restrict__ ssm,     // [L,SSM_LD]: B at 128..143, C at 144..159
                           const float* __restrict__ dtraw,   // [L,E]
                           const float* __restrict__ dt_b,    // [E]
                           const float* __restrict__ A_log,   // [E,16]
                           const float* __restrict__ Dp,      // [E]
                           const float* __restrict__ proj,    // [L,2E]: gate at col E+e
                           const float* __restrict__ h,       // [L,E] conv out
                           __bf16* __restrict__ ybf)          // [L,E] out
{
    int e = blockIdx.x * blockDim.x + threadIdx.x;
    float Ar[N_SSM], s[N_SSM];
    #pragma unroll
    for (int n = 0; n < N_SSM; ++n) {
        Ar[n] = -__expf(A_log[(size_t)e * N_SSM + n]);
        s[n] = 0.0f;
    }
    const float bias  = dt_b[e];
    const float dskip = Dp[e];

    for (int l = 0; l < L_SEQ; ++l) {
        float dtv = dtraw[(size_t)l * E_DIM + e] + bias;
        dtv = (dtv > 20.0f) ? dtv : log1pf(__expf(dtv));      // softplus
        float hv  = h[(size_t)l * E_DIM + e];
        float dth = dtv * hv;
        float y   = 0.0f;
        const float* Bc = ssm + (size_t)l * SSM_LD + R_DT;
        #pragma unroll
        for (int n = 0; n < N_SSM; ++n) {
            s[n] = fmaf(s[n], __expf(dtv * Ar[n]), dth * Bc[n]);
            y    = fmaf(s[n], Bc[N_SSM + n], y);
        }
        y = fmaf(hv, dskip, y);
        float g = proj[(size_t)l * (2 * E_DIM) + E_DIM + e];
        y *= g / (1.0f + __expf(-g));                          // * silu(gate)
        ybf[(size_t)l * E_DIM + e] = (__bf16)y;
    }
}

// ---------------------------------------------------------------------------
extern "C" void kernel_launch(void* const* d_in, const int* in_sizes, int n_in,
                              void* d_out, int out_size, void* d_ws, size_t ws_size,
                              hipStream_t stream)
{
    (void)in_sizes; (void)n_in; (void)out_size; (void)ws_size;

    const float* hs     = (const float*)d_in[0];  // [L,H]
    const float* w_in   = (const float*)d_in[1];  // [2E,H]
    const float* conv_w = (const float*)d_in[2];  // [E,1,4]
    const float* conv_b = (const float*)d_in[3];  // [E]
    const float* w_x    = (const float*)d_in[4];  // [160,E]
    const float* w_dt   = (const float*)d_in[5];  // [E,128]
    const float* b_dt   = (const float*)d_in[6];  // [E]
    const float* A_log  = (const float*)d_in[7];  // [E,16]
    const float* Dp     = (const float*)d_in[8];  // [E]
    const float* w_out  = (const float*)d_in[9];  // [H,E]
    float* out = (float*)d_out;                   // [L,H]

    char* ws = (char*)d_ws;
    size_t off = 0;
    float*  proj    = (float*) (ws + off); off += (size_t)L_SEQ * 2 * E_DIM * 4;   // 64MB
    float*  hbuf    = (float*) (ws + off); off += (size_t)L_SEQ * E_DIM * 4;       // 32MB
    float*  ssm     = (float*) (ws + off); off += (size_t)L_SEQ * SSM_LD * 4;      //  2MB
    float*  dtraw   = (float*) (ws + off); off += (size_t)L_SEQ * E_DIM * 4;       // 32MB
    __bf16* hs_bf   = (__bf16*)(ws + off); off += (size_t)L_SEQ * H_DIM * 2;       //  8MB
    __bf16* w_in_bf = (__bf16*)(ws + off); off += (size_t)2 * E_DIM * H_DIM * 2;   // 32MB
    __bf16* w_x_bf  = (__bf16*)(ws + off); off += (size_t)SSM_LD * E_DIM * 2;      //  2MB
    __bf16* w_dt_bf = (__bf16*)(ws + off); off += (size_t)E_DIM * R_DT * 2;        //  1MB
    __bf16* w_out_bf= (__bf16*)(ws + off); off += (size_t)H_DIM * E_DIM * 2;       // 16MB
    __bf16* ssm_bf  = (__bf16*)(ws + off); off += (size_t)L_SEQ * SSM_LD * 2;      //  1MB
    // w_in_bf region (32MB) is dead after GEMM1; recycle for h_bf + y_bf
    __bf16* h_bf = w_in_bf;
    __bf16* y_bf = w_in_bf + (size_t)L_SEQ * E_DIM;

    // --- 0) bf16 conversions (weights + input) ---
    {
        int n;
        n = L_SEQ * H_DIM;
        cvt_f32_bf16<<<n / 1024, 256, 0, stream>>>(hs, hs_bf, n);
        n = 2 * E_DIM * H_DIM;
        cvt_f32_bf16<<<n / 1024, 256, 0, stream>>>(w_in, w_in_bf, n);
        n = E_DIM * R_DT;
        cvt_f32_bf16<<<n / 1024, 256, 0, stream>>>(w_dt, w_dt_bf, n);
        n = H_DIM * E_DIM;
        cvt_f32_bf16<<<n / 1024, 256, 0, stream>>>(w_out, w_out_bf, n);
        padcvt_wx<<<dim3(E_DIM / 256, SSM_LD), 256, 0, stream>>>(w_x, w_x_bf);
    }

    dim3 gblk(256);

    // 1) in_proj: proj[L,2E] = hs * w_in^T
    gemm_bt_wmma<<<dim3(2 * E_DIM / 128, L_SEQ / 128), gblk, 0, stream>>>(
        hs_bf, H_DIM, w_in_bf, H_DIM, proj, 2 * E_DIM, H_DIM);

    // 2) depthwise causal conv + SiLU -> hbuf (f32) + h_bf (bf16)
    mamba_conv_silu<<<dim3(E_DIM / 256, L_SEQ), 256, 0, stream>>>(
        proj, conv_w, conv_b, hbuf, h_bf);

    // 3) x_proj: ssm[L,256] = h * w_x_pad^T
    gemm_bt_wmma<<<dim3(SSM_LD / 128, L_SEQ / 128), gblk, 0, stream>>>(
        h_bf, E_DIM, w_x_bf, E_DIM, ssm, SSM_LD, E_DIM);

    // 3b) ssm -> bf16 (dt columns feed the dt_proj GEMM)
    {
        int n = L_SEQ * SSM_LD;
        cvt_f32_bf16<<<n / 1024, 256, 0, stream>>>(ssm, ssm_bf, n);
    }

    // 4) dt_proj: dtraw[L,E] = ssm[:, :128] * w_dt^T   (lda = 256)
    gemm_bt_wmma<<<dim3(E_DIM / 128, L_SEQ / 128), gblk, 0, stream>>>(
        ssm_bf, SSM_LD, w_dt_bf, R_DT, dtraw, E_DIM, R_DT);

    // 5) selective scan -> y_bf
    mamba_scan<<<dim3(E_DIM / 256), 256, 0, stream>>>(
        ssm, dtraw, b_dt, A_log, Dp, proj, hbuf, y_bf);

    // 6) out_proj: out[L,H] = y * w_out^T
    gemm_bt_wmma<<<dim3(H_DIM / 128, L_SEQ / 128), gblk, 0, stream>>>(
        y_bf, E_DIM, w_out_bf, E_DIM, out, H_DIM, E_DIM);
}
